// HypergraphLayer_10677288698626
// MI455X (gfx1250) — compile-verified, matrix operand order checked by reading, not run on hardware
//
#include <hip/hip_runtime.h>

// ---------------------------------------------------------------------------
// HypergraphLayer fused CDNA5 (gfx1250).
// bf16 WMMA (v_wmma_f32_16x16x32_bf16) for all four GEMMs, fp32 accumulate.
// Weights pre-swizzled to fragment-major order (one 32B contiguous chunk per
// lane per fragment); LDS A tiles stored in the same swizzled layout so all
// fragment traffic is b128-class.
// ---------------------------------------------------------------------------

#define DN   256
#define DH   256
#define DHA  64
#define KC   (DHA + DH)      // 320
#define NNODES 50000
#define MINC   200000
#define EHYP   10000
#define TM   32              // incidence rows per block (2 waves * 16)

typedef __attribute__((ext_vector_type(16))) __bf16 v16bf;
typedef __attribute__((ext_vector_type(8)))  __bf16 v8bf;
typedef __attribute__((ext_vector_type(8)))  float  v8f;
typedef __attribute__((ext_vector_type(4)))  float  v4f;

#define WMMA_BF16(a, b, c) \
  __builtin_amdgcn_wmma_f32_16x16x32_bf16(false, (a), false, (b), (short)0, (c), false, false)

// ---------------------------------------------------------------------------
// Fragment swizzle helpers.
// CDNA5 16-bit A/B layout: lane = half*16 + idx16, element e in [0,16):
//   k(e,half) = (e&8 ? 16 : 0) + ((e>>1)&3)*2 + (e&1) + half*8
// Swizzled storage: element i = ((nt*numKt + kt)*32 + lane)*16 + e  (bf16 units)
// => each lane's fragment is 32 contiguous bytes.
// ---------------------------------------------------------------------------
__device__ __forceinline__ void swz_decode(int i, int numKt, int& k, int& n) {
  int e    = i & 15;
  int lane = (i >> 4) & 31;
  int frag = i >> 9;
  int kt = frag % numKt, nt = frag / numKt;
  int half = lane >> 4, idx16 = lane & 15;
  k = kt * 32 + ((e & 8) ? 16 : 0) + (((e >> 1) & 3) * 2) + (e & 1) + half * 8;
  n = nt * 16 + idx16;
}

// Octet property: 8 consecutive K-columns c0..c0+7 (c0 % 8 == 0) of one row map
// to 8 contiguous swizzled elements. Returns bf16-element offset of that chunk.
__device__ __forceinline__ int a_swz_base(int r, int c0, int numKt) {
  int rt = r >> 4, m = r & 15;
  int kt = c0 >> 5, kk = c0 & 31;
  int half  = (kk >> 3) & 1;
  int ebase = (kk & 16) ? 8 : 0;
  return (((rt * numKt + kt) * 32) + half * 16 + m) * 16 + ebase;
}

// ---------------------------------------------------------------------------
// K0: convert + swizzle all weight matrices fp32 -> bf16 fragment order.
// ---------------------------------------------------------------------------
__global__ void hg_cvt_weights(const float* __restrict__ w1, const float* __restrict__ w2,
                               const float* __restrict__ e1, const float* __restrict__ e2,
                               __bf16* __restrict__ o1, __bf16* __restrict__ o2,
                               __bf16* __restrict__ oe1, __bf16* __restrict__ oe2) {
  int i = blockIdx.x * 256 + threadIdx.x;      // grid covers 81920
  if (i < DN * DH) {
    int k, n;
    swz_decode(i, 8, k, n);
    o1[i]  = (__bf16)w1[k * DH + n];
    o2[i]  = (__bf16)w2[k * DH + n];
    oe2[i] = (__bf16)e2[k * DN + n];
  }
  if (i < KC * DH) {
    int k, n;
    swz_decode(i, 10, k, n);
    oe1[i] = (__bf16)e1[k * DH + n];
  }
}

// ---------------------------------------------------------------------------
// K1: node->edge fused MLP per 32-row incidence tile.
// ---------------------------------------------------------------------------
__global__ void hg_n2e(const float* __restrict__ x,
                       const long long* __restrict__ node_ids,
                       const long long* __restrict__ he_ids,
                       const __bf16* __restrict__ W1sw, const float* __restrict__ b1,
                       const float* __restrict__ lng, const float* __restrict__ lnb,
                       const __bf16* __restrict__ W2sw, const float* __restrict__ b2,
                       float* __restrict__ agg) {
  __shared__ __align__(16) char smem[TM * DN * sizeof(__bf16) + TM * DH * sizeof(float)];
  __bf16* sA = (__bf16*)smem;                               // 32x256 bf16, swizzled
  float*  sH = (float*)(smem + TM * DN * sizeof(__bf16));   // 32x256 f32 row-major

  const int tid  = threadIdx.x;          // 64 threads, 2 waves
  const int m0   = blockIdx.x * TM;
  const int lane = tid & 31, wave = tid >> 5;
  const int laneN = lane & 15, half = lane >> 4;

  __builtin_prefetch(W1sw + (size_t)tid * 1024, 0, 1);   // global_prefetch_b8
  __builtin_prefetch(W2sw + (size_t)tid * 1024, 0, 1);

  // ---- gather + fp32->bf16, store swizzled (b128 loads + b128 LDS stores) ----
  {
    const int OCT = DN / 8;              // 32 octets per row
    for (int idx = tid; idx < TM * OCT; idx += 64) {
      int r = idx / OCT, c0 = (idx % OCT) * 8;
      int m = m0 + r;
      v8bf val;
      if (m < MINC) {
        long long nid = node_ids[m];
        const float* p = x + (size_t)nid * DN + c0;
        v4f f0 = *(const v4f*)p;
        v4f f1 = *(const v4f*)(p + 4);
#pragma unroll
        for (int j = 0; j < 4; ++j) { val[j] = (__bf16)f0[j]; val[4 + j] = (__bf16)f1[j]; }
      } else {
#pragma unroll
        for (int j = 0; j < 8; ++j) val[j] = (__bf16)0.f;
      }
      *(v8bf*)(sA + a_swz_base(r, c0, 8)) = val;
    }
  }
  __syncthreads();

  // ---- GEMM1: h = A @ W1 + b1 -> sH ----
  const v16bf* sAv = (const v16bf*)sA;
  const v16bf* B1  = (const v16bf*)W1sw;
  const v16bf* B2  = (const v16bf*)W2sw;
  for (int nt = 0; nt < 16; ++nt) {
    v8f acc = {};
#pragma unroll
    for (int kt = 0; kt < 8; ++kt) {
      v16bf a = sAv[(wave * 8 + kt) * 32 + lane];
      v16bf b = B1[(nt * 8 + kt) * 32 + lane];
      acc = WMMA_BF16(a, b, acc);
    }
    int n = nt * 16 + laneN;
    float bias = b1[n];
#pragma unroll
    for (int r = 0; r < 8; ++r) {
      int mr = wave * 16 + r + 8 * half;
      sH[mr * DH + n] = acc[r] + bias;
    }
  }
  __syncthreads();

  // ---- LayerNorm + ReLU -> bf16 swizzled back into sA ----
  {
    int r = tid >> 1, h2 = tid & 1;
    const float* hp = sH + r * DH + h2 * 128;
    float s = 0.f, ss = 0.f;
#pragma unroll
    for (int q = 0; q < 32; ++q) {
      v4f v = *(const v4f*)(hp + q * 4);
#pragma unroll
      for (int j = 0; j < 4; ++j) { s += v[j]; ss += v[j] * v[j]; }
    }
    s += __shfl_xor(s, 1); ss += __shfl_xor(ss, 1);
    float mean = s * (1.f / DH);
    float var  = ss * (1.f / DH) - mean * mean;
    float rstd = rsqrtf(var + 1e-5f);
#pragma unroll
    for (int o = 0; o < 16; ++o) {
      int c0 = h2 * 128 + o * 8;
      v4f h0 = *(const v4f*)(sH + r * DH + c0);
      v4f h1 = *(const v4f*)(sH + r * DH + c0 + 4);
      v4f g0 = *(const v4f*)(lng + c0);
      v4f g1 = *(const v4f*)(lng + c0 + 4);
      v4f q0 = *(const v4f*)(lnb + c0);
      v4f q1 = *(const v4f*)(lnb + c0 + 4);
      v8bf val;
#pragma unroll
      for (int j = 0; j < 4; ++j) {
        val[j]     = (__bf16)fmaxf((h0[j] - mean) * rstd * g0[j] + q0[j], 0.f);
        val[4 + j] = (__bf16)fmaxf((h1[j] - mean) * rstd * g1[j] + q1[j], 0.f);
      }
      *(v8bf*)(sA + a_swz_base(r, c0, 8)) = val;
    }
  }
  __syncthreads();

  // ---- GEMM2: msgs = h @ W2 + b2 -> atomic scatter into agg_e ----
  for (int nt = 0; nt < 16; ++nt) {
    v8f acc = {};
#pragma unroll
    for (int kt = 0; kt < 8; ++kt) {
      v16bf a = sAv[(wave * 8 + kt) * 32 + lane];
      v16bf b = B2[(nt * 8 + kt) * 32 + lane];
      acc = WMMA_BF16(a, b, acc);
    }
    int n = nt * 16 + laneN;
    float bias = b2[n];
#pragma unroll
    for (int r = 0; r < 8; ++r) {
      int m = m0 + wave * 16 + r + 8 * half;
      if (m < MINC) {
        long long e = he_ids[m];
        atomicAdd(&agg[e * DH + n], acc[r] + bias);
      }
    }
  }
}

// ---------------------------------------------------------------------------
// K2: agg_e /= (count + eps)
// ---------------------------------------------------------------------------
__global__ void hg_norm_agg(float* __restrict__ agg, const float* __restrict__ cnt) {
  int i = blockIdx.x * 256 + threadIdx.x;      // E*DH = 2,560,000
  if (i < EHYP * DH) agg[i] = agg[i] / (cnt[i >> 8] + 1e-6f);
}

// ---------------------------------------------------------------------------
// K3: edge->node fused MLP per 32-row incidence tile.
// ---------------------------------------------------------------------------
__global__ void hg_e2n(const float* __restrict__ he_attr,
                       const float* __restrict__ agg,
                       const long long* __restrict__ he_ids,
                       const long long* __restrict__ node_ids,
                       const __bf16* __restrict__ W1sw, const float* __restrict__ b1,
                       const float* __restrict__ lng, const float* __restrict__ lnb,
                       const __bf16* __restrict__ W2sw, const float* __restrict__ b2,
                       float* __restrict__ accN, float* __restrict__ deg) {
  __shared__ __align__(16) char smem[TM * KC * sizeof(__bf16) + TM * DH * sizeof(float)];
  __bf16* sA = (__bf16*)smem;                               // 32x320 bf16 swizzled (numKt=10)
  float*  sH = (float*)(smem + TM * KC * sizeof(__bf16));   // 32x256 f32 row-major

  const int tid  = threadIdx.x;          // 64 threads, 2 waves
  const int m0   = blockIdx.x * TM;
  const int lane = tid & 31, wave = tid >> 5;
  const int laneN = lane & 15, half = lane >> 4;

  __builtin_prefetch(W1sw + (size_t)tid * 1280, 0, 1);
  __builtin_prefetch(W2sw + (size_t)tid * 1024, 0, 1);

  // ---- gather concat(he_attr, agg_e), swizzled bf16 ----
  {
    const int OCT = KC / 8;              // 40 octets per row
    for (int idx = tid; idx < TM * OCT; idx += 64) {
      int r = idx / OCT, c0 = (idx % OCT) * 8;
      int m = m0 + r;
      v8bf val;
      if (m < MINC) {
        long long e = he_ids[m];
        const float* p = (c0 < DHA) ? (he_attr + (size_t)e * DHA + c0)
                                    : (agg + (size_t)e * DH + (c0 - DHA));
        v4f f0 = *(const v4f*)p;
        v4f f1 = *(const v4f*)(p + 4);
#pragma unroll
        for (int j = 0; j < 4; ++j) { val[j] = (__bf16)f0[j]; val[4 + j] = (__bf16)f1[j]; }
      } else {
#pragma unroll
        for (int j = 0; j < 8; ++j) val[j] = (__bf16)0.f;
      }
      *(v8bf*)(sA + a_swz_base(r, c0, 10)) = val;
    }
  }
  __syncthreads();

  // ---- GEMM1: g = A @ E1 + b1 -> sH  (K = 320 -> 10 k-tiles) ----
  const v16bf* sAv = (const v16bf*)sA;
  const v16bf* B1  = (const v16bf*)W1sw;
  const v16bf* B2  = (const v16bf*)W2sw;
  for (int nt = 0; nt < 16; ++nt) {
    v8f acc = {};
#pragma unroll
    for (int kt = 0; kt < 10; ++kt) {
      v16bf a = sAv[(wave * 10 + kt) * 32 + lane];
      v16bf b = B1[(nt * 10 + kt) * 32 + lane];
      acc = WMMA_BF16(a, b, acc);
    }
    int n = nt * 16 + laneN;
    float bias = b1[n];
#pragma unroll
    for (int r = 0; r < 8; ++r) {
      int mr = wave * 16 + r + 8 * half;
      sH[mr * DH + n] = acc[r] + bias;
    }
  }
  __syncthreads();

  // ---- LayerNorm + ReLU -> bf16 swizzled, reuse sA (numKt=8 layout) ----
  {
    int r = tid >> 1, h2 = tid & 1;
    const float* hp = sH + r * DH + h2 * 128;
    float s = 0.f, ss = 0.f;
#pragma unroll
    for (int q = 0; q < 32; ++q) {
      v4f v = *(const v4f*)(hp + q * 4);
#pragma unroll
      for (int j = 0; j < 4; ++j) { s += v[j]; ss += v[j] * v[j]; }
    }
    s += __shfl_xor(s, 1); ss += __shfl_xor(ss, 1);
    float mean = s * (1.f / DH);
    float var  = ss * (1.f / DH) - mean * mean;
    float rstd = rsqrtf(var + 1e-5f);
#pragma unroll
    for (int o = 0; o < 16; ++o) {
      int c0 = h2 * 128 + o * 8;
      v4f h0 = *(const v4f*)(sH + r * DH + c0);
      v4f h1 = *(const v4f*)(sH + r * DH + c0 + 4);
      v4f g0 = *(const v4f*)(lng + c0);
      v4f g1 = *(const v4f*)(lng + c0 + 4);
      v4f q0 = *(const v4f*)(lnb + c0);
      v4f q1 = *(const v4f*)(lnb + c0 + 4);
      v8bf val;
#pragma unroll
      for (int j = 0; j < 4; ++j) {
        val[j]     = (__bf16)fmaxf((h0[j] - mean) * rstd * g0[j] + q0[j], 0.f);
        val[4 + j] = (__bf16)fmaxf((h1[j] - mean) * rstd * g1[j] + q1[j], 0.f);
      }
      *(v8bf*)(sA + a_swz_base(r, c0, 8)) = val;
    }
  }
  __syncthreads();

  // ---- GEMM2: inc_msg = relu(g @ E2 + b2) -> atomic scatter to nodes ----
  for (int nt = 0; nt < 16; ++nt) {
    v8f acc = {};
#pragma unroll
    for (int kt = 0; kt < 8; ++kt) {
      v16bf a = sAv[(wave * 8 + kt) * 32 + lane];
      v16bf b = B2[(nt * 8 + kt) * 32 + lane];
      acc = WMMA_BF16(a, b, acc);
    }
    int n = nt * 16 + laneN;
    float bias = b2[n];
#pragma unroll
    for (int r = 0; r < 8; ++r) {
      int m = m0 + wave * 16 + r + 8 * half;
      if (m < MINC) {
        long long nid = node_ids[m];
        float v = fmaxf(acc[r] + bias, 0.f);
        atomicAdd(&accN[nid * DN + n], v);
        if (nt == 0 && laneN == 0) atomicAdd(&deg[nid], 1.0f);
      }
    }
  }
}

// ---------------------------------------------------------------------------
// K4: out = x + LN(accN / (deg + eps)); one wave per node row.
// ---------------------------------------------------------------------------
__global__ void hg_finalize(const float* __restrict__ x,
                            const float* __restrict__ accN,
                            const float* __restrict__ deg,
                            const float* __restrict__ g, const float* __restrict__ b,
                            float* __restrict__ out) {
  int row = blockIdx.x;
  int lane = threadIdx.x;                 // 32 threads
  const float* a = accN + (size_t)row * DN;
  float d = 1.f / (deg[row] + 1e-6f);
  float v[8];
  float s = 0.f, ss = 0.f;
#pragma unroll
  for (int j = 0; j < 8; ++j) {
    v[j] = a[lane + 32 * j] * d;
    s += v[j]; ss += v[j] * v[j];
  }
#pragma unroll
  for (int off = 16; off > 0; off >>= 1) {
    s  += __shfl_xor(s, off);
    ss += __shfl_xor(ss, off);
  }
  float mean = s * (1.f / DN);
  float var  = ss * (1.f / DN) - mean * mean;
  float rstd = rsqrtf(var + 1e-5f);
#pragma unroll
  for (int j = 0; j < 8; ++j) {
    int c = lane + 32 * j;
    float y = (v[j] - mean) * rstd * g[c] + b[c];
    out[(size_t)row * DN + c] = x[(size_t)row * DN + c] + y;
  }
}

// ---------------------------------------------------------------------------
// Workspace layout (bytes)
// ---------------------------------------------------------------------------
static constexpr size_t OFF_W1  = 0;
static constexpr size_t OFF_W2  = OFF_W1  + (size_t)DN * DH * 2;
static constexpr size_t OFF_E1  = OFF_W2  + (size_t)DH * DH * 2;
static constexpr size_t OFF_E2  = OFF_E1  + (size_t)KC * DH * 2;
static constexpr size_t OFF_AGG = OFF_E2  + (size_t)DH * DN * 2;
static constexpr size_t OFF_ACC = OFF_AGG + (size_t)EHYP * DH * 4;
static constexpr size_t OFF_DEG = OFF_ACC + (size_t)NNODES * DN * 4;

extern "C" void kernel_launch(void* const* d_in, const int* in_sizes, int n_in,
                              void* d_out, int out_size, void* d_ws, size_t ws_size,
                              hipStream_t stream) {
  const float*     x        = (const float*)d_in[0];
  const long long* node_ids = (const long long*)d_in[1];
  const long long* he_ids   = (const long long*)d_in[2];
  const float*     he_attr  = (const float*)d_in[3];
  const float*     he_count = (const float*)d_in[4];
  const float*     n2e_w1   = (const float*)d_in[5];
  const float*     n2e_b1   = (const float*)d_in[6];
  const float*     n2e_lg   = (const float*)d_in[7];
  const float*     n2e_lb   = (const float*)d_in[8];
  const float*     n2e_w2   = (const float*)d_in[9];
  const float*     n2e_b2   = (const float*)d_in[10];
  const float*     e2n_w1   = (const float*)d_in[11];
  const float*     e2n_b1   = (const float*)d_in[12];
  const float*     e2n_lg   = (const float*)d_in[13];
  const float*     e2n_lb   = (const float*)d_in[14];
  const float*     e2n_w2   = (const float*)d_in[15];
  const float*     e2n_b2   = (const float*)d_in[16];
  const float*     ln_g     = (const float*)d_in[17];
  const float*     ln_b     = (const float*)d_in[18];
  float*           out      = (float*)d_out;

  char* ws = (char*)d_ws;
  __bf16* wb_w1 = (__bf16*)(ws + OFF_W1);
  __bf16* wb_w2 = (__bf16*)(ws + OFF_W2);
  __bf16* wb_e1 = (__bf16*)(ws + OFF_E1);
  __bf16* wb_e2 = (__bf16*)(ws + OFF_E2);
  float*  agg   = (float*)(ws + OFF_AGG);
  float*  accN  = (float*)(ws + OFF_ACC);
  float*  deg   = (float*)(ws + OFF_DEG);

  hipMemsetAsync(agg,  0, (size_t)EHYP * DH * sizeof(float),   stream);
  hipMemsetAsync(accN, 0, (size_t)NNODES * DN * sizeof(float), stream);
  hipMemsetAsync(deg,  0, (size_t)NNODES * sizeof(float),      stream);

  hg_cvt_weights<<<(KC * DH + 255) / 256, 256, 0, stream>>>(
      n2e_w1, n2e_w2, e2n_w1, e2n_w2, wb_w1, wb_w2, wb_e1, wb_e2);

  int nblk = (MINC + TM - 1) / TM;   // 6250
  hg_n2e<<<nblk, 64, 0, stream>>>(x, node_ids, he_ids,
                                  wb_w1, n2e_b1, n2e_lg, n2e_lb,
                                  wb_w2, n2e_b2, agg);

  hg_norm_agg<<<(EHYP * DH + 255) / 256, 256, 0, stream>>>(agg, he_count);

  hg_e2n<<<nblk, 64, 0, stream>>>(he_attr, agg, he_ids, node_ids,
                                  wb_e1, e2n_b1, e2n_lg, e2n_lb,
                                  wb_e2, e2n_b2, accN, deg);

  hg_finalize<<<NNODES, 32, 0, stream>>>(x, accN, deg, ln_g, ln_b, out);
}
